// WalkConv_1623497638433
// MI455X (gfx1250) — matrix operand order repaired and checked by reading, not compiled, not run on hardware
//
#include <hip/hip_runtime.h>

// ---------------------------------------------------------------------------
// WalkConv on CDNA5 (gfx1250): everything mapped onto v_wmma_f32_16x16x32_f16.
// wave32; 1 wave per tile-block for compute kernels.
// ---------------------------------------------------------------------------

typedef __attribute__((ext_vector_type(16))) _Float16 v16h;
typedef __attribute__((ext_vector_type(8)))  float    v8f;

#define NN    512
#define HID   16
#define EDIM  8
#define NE    8192
#define H4    64

// LDS staging tile 16 x 64 f32, padded leading dim (dwords) to avoid bank conflicts
#define LDS_LD 68

// ---- workspace layout (bytes) ----
#define OFF_WFRAG 0u
// fragment sub-offsets in halves (frag = 512 halves = 32 lanes x 16)
#define HOFF_W0 0
#define HOFF_W1 2048
#define HOFF_W2 6144
#define HOFF_W3 10240
#define HOFF_V0 26624
#define HOFF_V1 28672
#define HOFF_V2 32768
#define HOFF_V3 36864
#define HOFF_L  37888
#define OFF_FC   (131072u)                       // 8192*256 f16 = 4 MB
#define OFF_MTR1 (OFF_FC + 4194304u)             // 512*512*16 f32 = 16 MB
#define OFF_CSR  (OFF_MTR1 + 16777216u)
// CSR sub-offsets (bytes from OFF_CSR): counts[512]@0, offs[513]@2048, list[8192]@4352

__device__ __forceinline__ v8f wmma16(v16h a, v16h b, v8f c) {
  return __builtin_amdgcn_wmma_f32_16x16x32_f16(false, a, false, b, (short)0, c,
                                                false, false);
}

// B-fragment load: frag = kc*NC+nc, per-lane 16 contiguous halves (32B aligned)
__device__ __forceinline__ v16h load_B_frag(const _Float16* wf, int frag, int lane) {
  return *(const v16h*)(wf + (size_t)frag * 512 + lane * 16);
}

// C-frag (v8f tiles) -> LDS 16x(ntiles*16) f32, optional bias+relu.
// C layout: lanes0-15 VGPR r -> M=r; lanes16-31 VGPR r -> M=r+8; N = lane&15.
__device__ __forceinline__ void stage_C_to_lds(float* lds, const v8f* cc, int ntiles,
                                               const float* bias, bool relu, int lane) {
  const int l15 = lane & 15;
  const int mb = (lane < 16) ? 0 : 8;
  for (int nc = 0; nc < ntiles; ++nc) {
    float b = bias ? bias[nc * 16 + l15] : 0.0f;
#pragma unroll
    for (int r = 0; r < 8; ++r) {
      float v = cc[nc][r] + b;
      if (relu) v = fmaxf(v, 0.0f);
      lds[(mb + r) * LDS_LD + nc * 16 + l15] = v;
    }
  }
}

// A-fragment (16x32 f16 chunk, K base kbase) from LDS h[m][k] (ld=LDS_LD).
// A layout: lane<16: row M=lane, idx i -> K=kbase+i, idx 8+i -> K=kbase+16+i
//           lane>=16: row M=lane-16, idx i -> K=kbase+8+i, idx 8+i -> K=kbase+24+i
__device__ __forceinline__ v16h load_A_from_lds(const float* lds, int kbase, int kmax,
                                                int lane) {
  const int m = lane & 15;
  const int kh = (lane < 16) ? 0 : 8;
  v16h a;
#pragma unroll
  for (int i = 0; i < 8; ++i) {
    int k0 = kbase + kh + i;
    int k1 = kbase + 16 + kh + i;
    a[i]     = (k0 < kmax) ? (_Float16)lds[m * LDS_LD + k0] : (_Float16)0.0f;
    a[i + 8] = (k1 < kmax) ? (_Float16)lds[m * LDS_LD + k1] : (_Float16)0.0f;
  }
  return a;
}

// one 64->64 + bias + relu layer through LDS
__device__ __forceinline__ void layer_64_64(float* lds, const _Float16* wf,
                                            const float* bias, int lane) {
  v16h a0 = load_A_from_lds(lds, 0, 64, lane);
  v16h a1 = load_A_from_lds(lds, 32, 64, lane);
  v8f cc[4];
#pragma unroll
  for (int nc = 0; nc < 4; ++nc) {
    v8f c = {};
    c = wmma16(a0, load_B_frag(wf, 0 * 4 + nc, lane), c);
    c = wmma16(a1, load_B_frag(wf, 1 * 4 + nc, lane), c);
    cc[nc] = c;
  }
  __syncthreads();
  stage_C_to_lds(lds, cc, 4, bias, true, lane);
  __syncthreads();
}

// ---------------------------------------------------------------------------
// Pack a f32 row-major weight (fanin x fanout) into f16 B-fragments.
// blockIdx.x = kc*NC + nc, NC = fanout/16. Zero-pads K >= fanin.
// ---------------------------------------------------------------------------
__global__ void pack_weight(const float* __restrict__ W, int fanin, int fanout,
                            _Float16* __restrict__ dst) {
  const int NC = fanout >> 4;
  const int kc = blockIdx.x / NC;
  const int nc = blockIdx.x % NC;
  const int lane = threadIdx.x;
  const int l15 = lane & 15;
  const int khi = (lane >> 4) * 8;
#pragma unroll
  for (int i = 0; i < 16; ++i) {
    int K = kc * 32 + khi + (i < 8 ? i : i + 8);
    int Ncol = nc * 16 + l15;
    float v = (K < fanin) ? W[(size_t)K * fanout + Ncol] : 0.0f;
    dst[((size_t)blockIdx.x * 32 + lane) * 16 + i] = (_Float16)v;
  }
}

// ---------------------------------------------------------------------------
// Edge MLP: 16 edges per wave: 8 -> 64 -> 64 -> 64 -> 256, write F transposed
// Fc[e][n][k] (f16) so stage-2 B-fragments are contiguous per lane.
// Output tile nc of layer-3 == F row k=nc (since 256 = 16*16 tiled by 16).
// ---------------------------------------------------------------------------
__global__ void edge_mlp_kernel(const float* __restrict__ edge_attr,
                                const float* __restrict__ b0, const float* __restrict__ b1,
                                const float* __restrict__ b2, const float* __restrict__ b3,
                                const _Float16* __restrict__ wfrag,
                                _Float16* __restrict__ Fc) {
  __shared__ float lds[16 * LDS_LD];
  const int lane = threadIdx.x;
  const int ebase = blockIdx.x * 16;
  const int l15 = lane & 15;
  const int rb = (lane < 16) ? 0 : 8;

  // layer 0: A = edge_attr 16x8 (K 8..31 zero)
  v16h a = {};
  if (lane < 16) {
    const float* row = edge_attr + (size_t)(ebase + lane) * EDIM;
#pragma unroll
    for (int i = 0; i < 8; ++i) a[i] = (_Float16)row[i];
  }
  v8f cc[4];
#pragma unroll
  for (int nc = 0; nc < 4; ++nc) {
    v8f c = {};
    cc[nc] = wmma16(a, load_B_frag(wfrag + HOFF_W0, nc, lane), c);
  }
  __syncthreads();
  stage_C_to_lds(lds, cc, 4, b0, true, lane);
  __syncthreads();

  layer_64_64(lds, wfrag + HOFF_W1, b1, lane);
  layer_64_64(lds, wfrag + HOFF_W2, b2, lane);

  // layer 3: 64 -> 256, no relu; scatter to Fc[e][n][k] as f16
  v16h a0 = load_A_from_lds(lds, 0, 64, lane);
  v16h a1 = load_A_from_lds(lds, 32, 64, lane);
  for (int nc = 0; nc < 16; ++nc) {
    v8f c = {};
    c = wmma16(a0, load_B_frag(wfrag + HOFF_W3, 0 * 16 + nc, lane), c);
    c = wmma16(a1, load_B_frag(wfrag + HOFF_W3, 1 * 16 + nc, lane), c);
    float bb = b3[nc * 16 + l15];
#pragma unroll
    for (int r = 0; r < 8; ++r) {
      int e = ebase + rb + r;
      Fc[(size_t)e * 256 + l15 * 16 + nc] = (_Float16)(c[r] + bb);
    }
  }
}

// ---------------------------------------------------------------------------
// CSR build (deterministic)
// ---------------------------------------------------------------------------
__global__ void zero_ints(int* p, int n) {
  int i = blockIdx.x * blockDim.x + threadIdx.x;
  if (i < n) p[i] = 0;
}
__global__ void count_dst(const int* __restrict__ dst, int* __restrict__ counts, int ne) {
  int e = blockIdx.x * blockDim.x + threadIdx.x;
  if (e < ne) atomicAdd(&counts[dst[e]], 1);
}
__global__ void scan_serial(const int* __restrict__ counts, int* __restrict__ offs, int n) {
  if (blockIdx.x == 0 && threadIdx.x == 0) {
    int s = 0;
    for (int i = 0; i < n; ++i) { offs[i] = s; s += counts[i]; }
    offs[n] = s;
  }
}
// per-dst ordered fill: deterministic edge order -> deterministic FP sums
__global__ void fill_csr(const int* __restrict__ dst, const int* __restrict__ offs,
                         int* __restrict__ list, int ne, int n) {
  int k = blockIdx.x * blockDim.x + threadIdx.x;
  if (k >= n) return;
  int pos = offs[k];
  for (int e = 0; e < ne; ++e)
    if (dst[e] == k) list[pos++] = e;
}

// ---------------------------------------------------------------------------
// Message pass: block (k = dst column, itile = 16 rows of i).
// Two edges per WMMA: A = [mtr_tile(src0) | mtr_tile(src1)] 16x32,
//                     B = [F_e0 ; F_e1] 32x16 (from Fc, contiguous per lane).
// Applies diagonal zero and /HID, stores mtr1 f32.
// ---------------------------------------------------------------------------
__global__ void message_kernel(const float* __restrict__ mtr,
                               const int* __restrict__ src_idx,
                               const int* __restrict__ offs,
                               const int* __restrict__ list,
                               const _Float16* __restrict__ Fc,
                               float* __restrict__ mtr1) {
  const int lane = threadIdx.x;
  const int k = blockIdx.x;
  const int itile = blockIdx.y;
  const int l15 = lane & 15;
  const int rb = (lane < 16) ? 0 : 8;
  const int cb = (lane < 16) ? 0 : 8;   // unused: each lane loads all 16 hid channels
  const int irow = itile * 16 + l15;    // A row (M) handled by this lane

  const int beg = offs[k];
  const int end = offs[k + 1];
  v8f acc = {};

  for (int p = beg; p < end; p += 2) {
    int e0 = list[p];
    bool has1 = (p + 1 < end);
    int e1 = has1 ? list[p + 1] : e0;
    int s0 = src_idx[e0];
    int s1 = src_idx[e1];
    const float* ap = mtr + ((size_t)irow * NN + ((lane < 16) ? s0 : s1)) * HID;
    const bool use = (lane < 16) || has1;
    v16h a;
#pragma unroll
    for (int i = 0; i < 16; ++i) {
      a[i] = use ? (_Float16)ap[i] : (_Float16)0.0f;  // pad odd edge
    }
    int eb = (lane < 16) ? e0 : e1;
    v16h b = *(const v16h*)(Fc + (size_t)eb * 256 + l15 * 16);
    acc = wmma16(a, b, acc);
  }

  // zero diagonal (i == k) then scale 1/HID
  int m0 = k - itile * 16;
  if (m0 >= 0 && m0 < 16) {
#pragma unroll
    for (int r = 0; r < 8; ++r)
      if (rb + r == m0) acc[r] = 0.0f;
  }
#pragma unroll
  for (int r = 0; r < 8; ++r) acc[r] *= (1.0f / HID);

#pragma unroll
  for (int r = 0; r < 8; ++r)
    mtr1[((size_t)(itile * 16 + rb + r) * NN + k) * HID + l15] = acc[r];
}

// ---------------------------------------------------------------------------
// Node stage: block (k, itile). X = mtr1[i,k,:] + mtr1[k,i,:]@L + Lb, then
// MLP 16->64->64->64->16 (WMMA), out = mtr + result.
// ---------------------------------------------------------------------------
__global__ void node_kernel(const float* __restrict__ mtr,
                            const float* __restrict__ mtr1,
                            const float* __restrict__ c0, const float* __restrict__ c1,
                            const float* __restrict__ c2, const float* __restrict__ c3,
                            const float* __restrict__ Lb,
                            const _Float16* __restrict__ wfrag,
                            float* __restrict__ out) {
  __shared__ float lds[16 * LDS_LD];
  const int lane = threadIdx.x;
  const int k = blockIdx.x;
  const int itile = blockIdx.y;
  const int l15 = lane & 15;
  const int rb = (lane < 16) ? 0 : 8;

  // A = mtr1[k, i, :] (pixel rows p, K = c; K>=16 zero).  Row k block is contiguous.
  v16h a = {};
  {
    const int cbase = (lane < 16) ? 0 : 8;
    const float* tp = mtr1 + ((size_t)k * NN + itile * 16 + l15) * HID + cbase;
#pragma unroll
    for (int i = 0; i < 8; ++i) a[i] = (_Float16)tp[i];
  }
  v16h bL = load_B_frag(wfrag + HOFF_L, 0, lane);

  // C init = mtr1[i,k,:] + Lb
  v8f x = {};
  {
    float lb = Lb[l15];
#pragma unroll
    for (int r = 0; r < 8; ++r)
      x[r] = mtr1[((size_t)(itile * 16 + rb + r) * NN + k) * HID + l15] + lb;
  }
  x = wmma16(a, bL, x);            // X = mtr1 + trans  (16x16, C layout)

  __syncthreads();
  stage_C_to_lds(lds, &x, 1, nullptr, false, lane);
  __syncthreads();

  // layer V0: 16 -> 64
  {
    v16h a0 = load_A_from_lds(lds, 0, 16, lane);
    v8f cc[4];
#pragma unroll
    for (int nc = 0; nc < 4; ++nc) {
      v8f c = {};
      cc[nc] = wmma16(a0, load_B_frag(wfrag + HOFF_V0, nc, lane), c);
    }
    __syncthreads();
    stage_C_to_lds(lds, cc, 4, c0, true, lane);
    __syncthreads();
  }
  layer_64_64(lds, wfrag + HOFF_V1, c1, lane);
  layer_64_64(lds, wfrag + HOFF_V2, c2, lane);

  // final V3: 64 -> 16, + c3, + residual mtr
  v16h f0 = load_A_from_lds(lds, 0, 64, lane);
  v16h f1 = load_A_from_lds(lds, 32, 64, lane);
  v8f d = {};
  d = wmma16(f0, load_B_frag(wfrag + HOFF_V3, 0, lane), d);
  d = wmma16(f1, load_B_frag(wfrag + HOFF_V3, 1, lane), d);
  float bb = c3[l15];
#pragma unroll
  for (int r = 0; r < 8; ++r) {
    size_t idx = ((size_t)(itile * 16 + rb + r) * NN + k) * HID + l15;
    out[idx] = mtr[idx] + d[r] + bb;
  }
}

// ---------------------------------------------------------------------------
extern "C" void kernel_launch(void* const* d_in, const int* in_sizes, int n_in,
                              void* d_out, int out_size, void* d_ws, size_t ws_size,
                              hipStream_t stream) {
  (void)in_sizes; (void)n_in; (void)out_size; (void)ws_size;
  const float* mtr       = (const float*)d_in[0];
  const float* edge_attr = (const float*)d_in[1];
  const int*   eidx      = (const int*)d_in[2];
  const float* W0 = (const float*)d_in[3];  const float* b0 = (const float*)d_in[4];
  const float* W1 = (const float*)d_in[5];  const float* b1 = (const float*)d_in[6];
  const float* W2 = (const float*)d_in[7];  const float* b2 = (const float*)d_in[8];
  const float* W3 = (const float*)d_in[9];  const float* b3 = (const float*)d_in[10];
  const float* V0 = (const float*)d_in[11]; const float* c0 = (const float*)d_in[12];
  const float* V1 = (const float*)d_in[13]; const float* c1 = (const float*)d_in[14];
  const float* V2 = (const float*)d_in[15]; const float* c2 = (const float*)d_in[16];
  const float* V3 = (const float*)d_in[17]; const float* c3 = (const float*)d_in[18];
  const float* L  = (const float*)d_in[19]; const float* Lb = (const float*)d_in[20];

  char* ws = (char*)d_ws;
  _Float16* wfrag = (_Float16*)(ws + OFF_WFRAG);
  _Float16* Fc    = (_Float16*)(ws + OFF_FC);
  float*    mtr1  = (float*)(ws + OFF_MTR1);
  int* counts = (int*)(ws + OFF_CSR);
  int* offs   = (int*)(ws + OFF_CSR + 2048);
  int* list   = (int*)(ws + OFF_CSR + 4352);
  const int* src = eidx;        // edge_index[0]
  const int* dst = eidx + NE;   // edge_index[1]

  // 1) pack weights to f16 B-fragments
  pack_weight<<<4,  32, 0, stream>>>(W0, EDIM, H4,        wfrag + HOFF_W0);
  pack_weight<<<8,  32, 0, stream>>>(W1, H4,   H4,        wfrag + HOFF_W1);
  pack_weight<<<8,  32, 0, stream>>>(W2, H4,   H4,        wfrag + HOFF_W2);
  pack_weight<<<32, 32, 0, stream>>>(W3, H4,   HID * HID, wfrag + HOFF_W3);
  pack_weight<<<4,  32, 0, stream>>>(V0, HID,  H4,        wfrag + HOFF_V0);
  pack_weight<<<8,  32, 0, stream>>>(V1, H4,   H4,        wfrag + HOFF_V1);
  pack_weight<<<8,  32, 0, stream>>>(V2, H4,   H4,        wfrag + HOFF_V2);
  pack_weight<<<2,  32, 0, stream>>>(V3, H4,   HID,       wfrag + HOFF_V3);
  pack_weight<<<1,  32, 0, stream>>>(L,  HID,  HID,       wfrag + HOFF_L);

  // 2) edge MLP -> Fc
  edge_mlp_kernel<<<NE / 16, 32, 0, stream>>>(edge_attr, b0, b1, b2, b3, wfrag, Fc);

  // 3) deterministic CSR by dst
  zero_ints<<<2, 256, 0, stream>>>(counts, NN);
  count_dst<<<NE / 256, 256, 0, stream>>>(dst, counts, NE);
  scan_serial<<<1, 1, 0, stream>>>(counts, offs, NN);
  fill_csr<<<2, 256, 0, stream>>>(dst, offs, list, NE, NN);

  // 4) message pass -> mtr1
  message_kernel<<<dim3(NN, NN / 16), 32, 0, stream>>>(mtr, src, offs, list, Fc, mtr1);

  // 5) transpose-linear + node MLP + residual -> out
  node_kernel<<<dim3(NN, NN / 16), 32, 0, stream>>>(mtr, mtr1, c0, c1, c2, c3, Lb,
                                                    wfrag, (float*)d_out);
}
